// ContrastiveLoss_18957985645125
// MI455X (gfx1250) — matrix operand order repaired and checked by reading, not compiled, not run on hardware
//
#include <hip/hip_runtime.h>
#include <hip/hip_bf16.h>

// ---------------- problem constants (from reference setup_inputs) -----------
constexpr int N_  = 4;
constexpr int E_  = 16;
constexpr int Hh  = 768;
constexpr int Ww  = 768;
constexpr int C_  = 32;
constexpr int HWP = Hh * Ww;              // 589824 pixels per sample

constexpr float DELTA_VAR  = 0.5f;
constexpr float DELTA_DIST = 2.0f;
constexpr float ALPHA = 1.0f, BETA = 1.0f, GAMMA = 0.001f;
constexpr float EPSf = 1e-12f;

// kernel-1: 288 blocks/sample * 4 waves * 4 tiles * 128 px = HWP
constexpr int BPS1   = 288;
constexpr int WAVES1 = 4;                            // 128 threads/block
constexpr int WPS1   = BPS1 * WAVES1;                // 1152 waves per sample
constexpr int TPW1   = HWP / (WPS1 * 128);           // 4 tiles per wave
constexpr int STR    = 132;                          // halfs per channel row (pad)
// kernel-3: 144 blocks/sample * 256 thr * 4 quads * 4 px = HWP
constexpr int BPS3 = 144;

typedef __attribute__((ext_vector_type(16))) _Float16 v16h;
typedef __attribute__((ext_vector_type(8)))  float    v8f;
typedef __attribute__((ext_vector_type(4)))  float    f4;
typedef __attribute__((ext_vector_type(4)))  _Float16 h4;
typedef __attribute__((ext_vector_type(4)))  int      i4;
typedef __attribute__((ext_vector_type(4)))  unsigned u4;

// ---------------------------------------------------------------------------
__global__ void k0_zero(float* __restrict__ ws, int n) {
  int i = blockIdx.x * 256 + threadIdx.x;
  if (i < n) ws[i] = 0.0f;
}

// ---------------------------------------------------------------------------
// k1: per-class segment sums via WMMA: sums = onehot(C x P) * emb(P x E).
//   emb split into f16 hi+lo (products exact vs 0/1 one-hot; f32 accumulate).
//   128-pixel tiles per wave, b128 streaming loads, LDS-staged fragments,
//   block-level LDS reduction before global atomics.
__global__ __launch_bounds__(128) void k1_segsum(const float* __restrict__ inp,
                                                 const int*   __restrict__ lab,
                                                 float* __restrict__ sums,
                                                 float* __restrict__ counts) {
  __shared__ _Float16 s_hi[WAVES1][16 * STR + 4];
  __shared__ _Float16 s_lo[WAVES1][16 * STR + 4];
  __shared__ unsigned s_lab[WAVES1][128];
  __shared__ float    s_cnt[C_];
  __shared__ float    s_red[WAVES1][C_ * E_];

  const int tid  = threadIdx.x;
  const int wave = tid >> 5;
  const int lane = tid & 31;
  const int n    = blockIdx.x / BPS1;
  const int bis  = blockIdx.x % BPS1;
  const int wsid = bis * WAVES1 + wave;              // wave id within sample

  for (int i = tid; i < C_; i += 128) s_cnt[i] = 0.0f;
  __syncthreads();

  const int nIdx = lane & 15;                        // B column / class-in-tile
  const int hsel = lane >> 4;                        // lane half
  const float* __restrict__ base = inp + (size_t)n * E_ * HWP;
  const int*   __restrict__ labn = lab + (size_t)n * HWP;

  v8f acc0 = {};   // classes 0..15
  v8f acc1 = {};   // classes 16..31

  for (int t = 0; t < TPW1; ++t) {
    const int p0 = (wsid + t * WPS1) * 128;
    if (t + 1 < TPW1) {                              // global_prefetch_b8
      const int pn = (wsid + (t + 1) * WPS1) * 128;
      __builtin_prefetch(base + pn + 4 * lane, 0, 3);
      __builtin_prefetch(labn + pn + 4 * lane, 0, 3);
    }
    // 128 labels per wave: one b128 per lane
    const i4 l4 = *(const i4*)&labn[p0 + 4 * lane];
    *(u4*)&s_lab[wave][4 * lane] = (u4)l4;
    atomicAdd(&s_cnt[l4.x], 1.0f);
    atomicAdd(&s_cnt[l4.y], 1.0f);
    atomicAdd(&s_cnt[l4.z], 1.0f);
    atomicAdd(&s_cnt[l4.w], 1.0f);
    // 16 channel rows, float4 per lane -> hi/lo f16 split -> LDS
    #pragma unroll
    for (int e = 0; e < 16; ++e) {
      f4 x  = *(const f4*)&base[(size_t)e * HWP + p0 + 4 * lane];
      h4 hi = __builtin_convertvector(x, h4);
      f4 rm = x - __builtin_convertvector(hi, f4);
      h4 lo = __builtin_convertvector(rm, h4);
      *(h4*)&s_hi[wave][e * STR + 4 * lane] = hi;
      *(h4*)&s_lo[wave][e * STR + 4 * lane] = lo;
    }
    __builtin_amdgcn_wave_barrier();
    asm volatile("" ::: "memory");

    // 4 K-blocks of 32 pixels each
    #pragma unroll
    for (int kblk = 0; kblk < 4; ++kblk) {
      const int kb0 = kblk * 32;
      union { v16h h; unsigned u[8]; } a0, a1, bh, bl;
      #pragma unroll
      for (int v = 0; v < 8; ++v) {
        // A 16x32 f16 layout: K = (v<4?0:16) + 8*hsel + 2*(v&3) (+1 packed)
        const int k  = kb0 + ((v < 4) ? 0 : 16) + 8 * hsel + ((v & 3) << 1);
        const unsigned l0 = s_lab[wave][k];
        const unsigned l1 = s_lab[wave][k + 1];
        const unsigned c0 = (unsigned)nIdx, c1 = (unsigned)(nIdx + 16);
        a0.u[v] = (l0 == c0 ? 0x3C00u : 0u) | ((l1 == c0 ? 0x3C00u : 0u) << 16);
        a1.u[v] = (l0 == c1 ? 0x3C00u : 0u) | ((l1 == c1 ? 0x3C00u : 0u) << 16);
        // B 32x16 f16 layout: K = (v<4?0:16) + 8*hsel + 2*(v&3) (+1), N = nIdx
        const int kb = kb0 + ((v < 4) ? 0 : 16) + 8 * hsel + ((v & 3) << 1);
        bh.u[v] = *(const unsigned*)&s_hi[wave][nIdx * STR + kb];
        bl.u[v] = *(const unsigned*)&s_lo[wave][nIdx * STR + kb];
      }
      acc0 = __builtin_amdgcn_wmma_f32_16x16x32_f16(false, a0.h, false, bh.h, (short)0, acc0, false, false);
      acc0 = __builtin_amdgcn_wmma_f32_16x16x32_f16(false, a0.h, false, bl.h, (short)0, acc0, false, false);
      acc1 = __builtin_amdgcn_wmma_f32_16x16x32_f16(false, a1.h, false, bh.h, (short)0, acc1, false, false);
      acc1 = __builtin_amdgcn_wmma_f32_16x16x32_f16(false, a1.h, false, bl.h, (short)0, acc1, false, false);
    }
    __builtin_amdgcn_wave_barrier();
  }

  // D layout: VGPR r -> M = r (lanes 0-15) / r+8 (lanes 16-31); N = nIdx
  #pragma unroll
  for (int r = 0; r < 8; ++r) {
    const int m = r + (hsel ? 8 : 0);
    s_red[wave][m * E_ + nIdx]        = acc0[r];
    s_red[wave][(16 + m) * E_ + nIdx] = acc1[r];
  }
  __syncthreads();
  // block-level reduction, then one atomic per (class, channel)
  float* __restrict__ sn = sums + (size_t)n * C_ * E_;
  for (int i = tid; i < C_ * E_; i += 128) {
    float v = s_red[0][i] + s_red[1][i] + s_red[2][i] + s_red[3][i];
    atomicAdd(&sn[i], v);
  }
  if (tid < C_) atomicAdd(&counts[n * C_ + tid], s_cnt[tid]);
}

// ---------------------------------------------------------------------------
// k2: means = sums/counts; distance + regularizer terms (one block/sample)
__global__ __launch_bounds__(256) void k2_means(const float* __restrict__ sums,
                                                const float* __restrict__ counts,
                                                float* __restrict__ means,
                                                float* __restrict__ partial) {
  __shared__ float m[C_ * E_];
  __shared__ float red[2];
  const int n = blockIdx.x, tid = threadIdx.x;
  if (tid < 2) red[tid] = 0.0f;
  __syncthreads();
  for (int i = tid; i < C_ * E_; i += 256) {
    float mm = sums[n * C_ * E_ + i] / counts[n * C_ + i / E_];
    m[i] = mm;
    means[n * C_ * E_ + i] = mm;
  }
  __syncthreads();
  if (tid < C_) {                               // regularizer: mean norms
    float s = 0.0f;
    #pragma unroll
    for (int e = 0; e < E_; ++e) { float v = m[tid * E_ + e]; s += v * v; }
    atomicAdd(&red[0], sqrtf(fmaxf(s, EPSf)));
  }
  float dsum = 0.0f;                            // pairwise hinged repulsion
  for (int idx = tid; idx < C_ * C_; idx += 256) {
    const int i = idx >> 5, j = idx & 31;
    if (i != j) {
      float s = 0.0f;
      #pragma unroll
      for (int e = 0; e < E_; ++e) {
        float d = m[i * E_ + e] - m[j * E_ + e];
        s += d * d;
      }
      float dist = sqrtf(fmaxf(s, EPSf));
      float hg = fmaxf(2.0f * DELTA_DIST - dist, 0.0f);
      dsum += hg * hg;
    }
  }
  atomicAdd(&red[1], dsum);
  __syncthreads();
  if (tid == 0) {
    float reg = red[0] / (float)C_;
    float dt  = red[1] / (float)(C_ * (C_ - 1));
    partial[n] = BETA * dt + GAMMA * reg;
  }
}

// ---------------------------------------------------------------------------
// k3: variance term — stream pixel quads (b128), hinge vs own cluster mean
__global__ __launch_bounds__(256) void k3_var(const float* __restrict__ inp,
                                              const int*   __restrict__ lab,
                                              const float* __restrict__ means,
                                              float* __restrict__ var_out) {
  __shared__ float m[C_ * E_];
  __shared__ float s_var[C_];
  const int tid = threadIdx.x;
  const int n = blockIdx.x / BPS3, b = blockIdx.x % BPS3;
  for (int i = tid; i < C_ * E_; i += 256) m[i] = means[n * C_ * E_ + i];
  for (int i = tid; i < C_; i += 256) s_var[i] = 0.0f;
  __syncthreads();
  const float* __restrict__ basep = inp + (size_t)n * E_ * HWP;
  const int*   __restrict__ labn  = lab + (size_t)n * HWP;
  const int qstride = BPS3 * 256;
  for (int q = b * 256 + tid; q < HWP / 4; q += qstride) {
    const int p = q * 4;
    const i4 lv = *(const i4*)&labn[p];
    float s0 = 0.f, s1 = 0.f, s2 = 0.f, s3 = 0.f;
    #pragma unroll
    for (int e = 0; e < 16; ++e) {
      f4 x = *(const f4*)&basep[(size_t)e * HWP + p];
      float d0 = x.x - m[lv.x * E_ + e]; s0 += d0 * d0;
      float d1 = x.y - m[lv.y * E_ + e]; s1 += d1 * d1;
      float d2 = x.z - m[lv.z * E_ + e]; s2 += d2 * d2;
      float d3 = x.w - m[lv.w * E_ + e]; s3 += d3 * d3;
    }
    float h0 = fmaxf(sqrtf(fmaxf(s0, EPSf)) - DELTA_VAR, 0.f);
    float h1 = fmaxf(sqrtf(fmaxf(s1, EPSf)) - DELTA_VAR, 0.f);
    float h2 = fmaxf(sqrtf(fmaxf(s2, EPSf)) - DELTA_VAR, 0.f);
    float h3 = fmaxf(sqrtf(fmaxf(s3, EPSf)) - DELTA_VAR, 0.f);
    atomicAdd(&s_var[lv.x], h0 * h0);
    atomicAdd(&s_var[lv.y], h1 * h1);
    atomicAdd(&s_var[lv.z], h2 * h2);
    atomicAdd(&s_var[lv.w], h3 * h3);
  }
  __syncthreads();
  if (tid < C_) atomicAdd(&var_out[n * C_ + tid], s_var[tid]);
}

// ---------------------------------------------------------------------------
__global__ __launch_bounds__(128) void k4_final(const float* __restrict__ var_s,
                                                const float* __restrict__ counts,
                                                const float* __restrict__ partial,
                                                float* __restrict__ out) {
  __shared__ float red;
  const int tid = threadIdx.x;                  // 128 == N_*C_
  if (tid == 0) red = 0.0f;
  __syncthreads();
  float v = ALPHA * (var_s[tid] / counts[tid]) / (float)C_;
  if (tid < N_) v += partial[tid];
  atomicAdd(&red, v);
  __syncthreads();
  if (tid == 0) out[0] = red / (float)N_;
}

// ---------------------------------------------------------------------------
extern "C" void kernel_launch(void* const* d_in, const int* in_sizes, int n_in,
                              void* d_out, int out_size, void* d_ws, size_t ws_size,
                              hipStream_t stream) {
  (void)in_sizes; (void)n_in; (void)out_size; (void)ws_size;
  const float* inp = (const float*)d_in[0];
  const int*   lab = (const int*)d_in[1];      // jnp int64 -> int32 (x64 off)

  float* ws      = (float*)d_ws;
  float* sums    = ws;                          // N*C*E = 2048
  float* counts  = ws + 2048;                   // N*C   = 128
  float* means   = ws + 2176;                   // N*C*E = 2048
  float* var_s   = ws + 4224;                   // N*C   = 128
  float* partial = ws + 4352;                   // N     = 4
  const int ztot = 4356;

  k0_zero  <<<(ztot + 255) / 256, 256, 0, stream>>>(ws, ztot);
  k1_segsum<<<N_ * BPS1, 128, 0, stream>>>(inp, lab, sums, counts);
  k2_means <<<N_, 256, 0, stream>>>(sums, counts, means, partial);
  k3_var   <<<N_ * BPS3, 256, 0, stream>>>(inp, lab, means, var_s);
  k4_final <<<1, 128, 0, stream>>>(var_s, counts, partial, (float*)d_out);
}
